// WrappedGLMBlock_33835752357900
// MI455X (gfx1250) — matrix operand order repaired and checked by reading, not compiled, not run on hardware
//
#include <hip/hip_runtime.h>

// ---------------- constants ----------------
#define SEQ    2048
#define HEADS  32
#define DHEAD  128
#define MODELD 4096
#define FFD    16384
#define RESC   7.483314773547883f   // sqrt(2*28)
#define QSCALE 0.08838834764831845f // 1/sqrt(128), coeff = 1

// ---------------- WMMA plumbing ----------------
typedef __attribute__((ext_vector_type(16))) __bf16 v16bf;
typedef __attribute__((ext_vector_type(8)))  float  v8f;
typedef __attribute__((ext_vector_type(4)))  unsigned int v4u;
typedef __attribute__((ext_vector_type(8)))  int v8i;
typedef __attribute__((ext_vector_type(4)))  int v4i;

union BFrag { v16bf v; unsigned int u[8]; };

__device__ __forceinline__ v8f wmma_bf16(const v16bf& a, const v16bf& b, const v8f& c) {
  return __builtin_amdgcn_wmma_f32_16x16x32_bf16(false, a, false, b, (short)0, c,
                                                 false, false);
}

__device__ __forceinline__ unsigned short f2bf(float x) {
  unsigned int u = __float_as_uint(x);
  unsigned int r = u + 0x7FFFu + ((u >> 16) & 1u);   // round-to-nearest-even
  return (unsigned short)(r >> 16);
}
__device__ __forceinline__ unsigned int pack2bf(float a, float b) {
  return (unsigned int)f2bf(a) | ((unsigned int)f2bf(b) << 16);
}

// async copy of 16B global -> LDS (GVS mode), tracked by ASYNCcnt
__device__ __forceinline__ void async_b128(unsigned lds_off, unsigned byte_off,
                                           const void* base) {
  asm volatile("global_load_async_to_lds_b128 %0, %1, %2"
               :: "v"(lds_off), "v"(byte_off), "s"(base) : "memory");
}

// ---- Tensor Data Mover: 2-D bf16 tile (rows x 32) -> LDS, rows padded 64B->80B ----
// D# per CDNA5 ISA: group0 {count=1, lds_addr, global_addr, type=2},
// group1 {data_size=2B, pad_enable, pad_interval=16dw, pad_amount=4dw,
//         tensor_dim0=K, tensor_dim1=NR, tile_dim0=32, tile_dim1=rows, dim0_stride=K}
__device__ __forceinline__ void tdm_tile_2d(const void* base, long long elem_off,
                                            int K_, int NR_, int rows,
                                            unsigned lds_addr) {
  unsigned long long ga = (unsigned long long)base + (unsigned long long)elem_off * 2ull;
  v4u g0;
  g0[0] = 1u;                                                    // count=1, user mode
  g0[1] = lds_addr;                                              // LDS byte address
  g0[2] = (unsigned)(ga & 0xFFFFFFFFull);                        // global_addr[31:0]
  g0[3] = (unsigned)((ga >> 32) & 0x01FFFFFFull) | (2u << 30);   // addr[56:32], type=2
  v8i g1;
  g1[0] = (int)((1u << 16) | (1u << 20) | (3u << 22) | (3u << 25));
  g1[1] = (int)(((unsigned)K_ & 0xFFFFu) << 16);                 // tensor_dim0 lo16
  g1[2] = (int)(((unsigned)K_ >> 16) | (((unsigned)NR_ & 0xFFFFu) << 16));
  g1[3] = (int)(((unsigned)NR_ >> 16) | (32u << 16));            // tile_dim0 = 32
  g1[4] = (int)(unsigned)rows;                                   // tile_dim1
  g1[5] = (int)(unsigned)K_;                                     // dim0_stride[31:0]
  g1[6] = 0;
  g1[7] = 0;
  v4i z = {};
#if __clang_major__ >= 23
  v8i z8 = {};
  __builtin_amdgcn_tensor_load_to_lds(g0, g1, z, z, z8, 0);
#else
  __builtin_amdgcn_tensor_load_to_lds(g0, g1, z, z, 0);
#endif
}

// ---------------- f32 -> bf16 conversion (optional scale) ----------------
__global__ __launch_bounds__(256)
void cvt_bf16_kernel(const float* __restrict__ in, unsigned short* __restrict__ out,
                     size_t n4, float scale) {
  size_t i = ((size_t)blockIdx.x * 256 + threadIdx.x);
  if (i < n4) {
    float4 x = *(const float4*)(in + i * 4);
    unsigned int* o = (unsigned int*)(out + i * 4);
    o[0] = pack2bf(x.x * scale, x.y * scale);
    o[1] = pack2bf(x.z * scale, x.w * scale);
  }
}

// v [S][H*D] f32  ->  vt [H][D][S] bf16 (transposed so attention tiles are contiguous)
__global__ __launch_bounds__(256)
void cvt_vt_kernel(const float* __restrict__ v, unsigned short* __restrict__ vt) {
  size_t i = (size_t)blockIdx.x * 256 + threadIdx.x;   // = (h*128 + d)*2048 + s
  int s = (int)(i & (SEQ - 1));
  size_t hd = i >> 11;
  vt[i] = f2bf(v[(size_t)s * MODELD + hd]);
}

// ---------------- flash attention: 1 wave per (16-query tile, head) ----------------
// qh: [S][MODELD] bf16 pre-scaled; kh: [S][MODELD] bf16; vt: [H][D][S] bf16
__global__ __launch_bounds__(32)
void attn_kernel(const unsigned short* __restrict__ qh,
                 const unsigned short* __restrict__ kh,
                 const unsigned short* __restrict__ vt,
                 unsigned short* __restrict__ ctx_h) {
  __shared__ unsigned short lds_k[32][136];    // [key][d] bf16, 272B rows (16B mult)
  __shared__ unsigned short lds_vt[DHEAD][40]; // [d][key] bf16, 80B rows (16B mult)
  __shared__ unsigned short lds_p[16][32];     // P tile bf16

  const int lane  = threadIdx.x;
  const int q0    = blockIdx.x * 16;
  const int head  = blockIdx.y;
  const int row16 = lane & 15;
  const int hb    = lane >> 4;

  const unsigned ldsK = (unsigned)(size_t)&lds_k[0][0];
  const unsigned ldsV = (unsigned)(size_t)&lds_vt[0][0];

  // Q fragments (A layout, 16x32 bf16 each)
  BFrag qf[4];
  {
    const unsigned short* qrow =
        qh + (size_t)(q0 + row16) * MODELD + head * DHEAD;
#pragma unroll
    for (int kk = 0; kk < 4; ++kk)
#pragma unroll
      for (int vg = 0; vg < 8; ++vg) {
        int d = kk * 32 + ((vg >= 4) ? 16 : 0) + hb * 8 + (vg & 3) * 2;
        qf[kk].u[vg] = *(const unsigned int*)&qrow[d];
      }
  }

  v8f ctx_acc[8];
#pragma unroll
  for (int t = 0; t < 8; ++t) ctx_acc[t] = {};
  float m_prev[8], l_run[8];
#pragma unroll
  for (int i = 0; i < 8; ++i) { m_prev[i] = -1e30f; l_run[i] = 0.0f; }

  const int nblocks = (q0 + 16 + 31) / 32;   // causal: keys <= q0+15
  for (int jb = 0; jb < nblocks; ++jb) {
    const int key0 = jb * 32;

    // --- async stage K tile [32 keys][128 d] : 16 b128 chunks per lane ---
    {
      unsigned kvoff = (unsigned)((((size_t)(key0 + lane)) * MODELD + head * DHEAD) * 2);
      unsigned kdst  = ldsK + lane * 272;
#pragma unroll
      for (int c = 0; c < 16; ++c)
        async_b128(kdst + c * 16, kvoff + c * 16, kh);
    }
    // --- async stage Vt tile [128 d][32 keys] : 16 chunks per lane ---
    {
#pragma unroll
      for (int c = 0; c < 16; ++c) {
        int id = lane * 16 + c;
        int dr = id >> 2, kc = id & 3;
        unsigned voff =
            (unsigned)((((size_t)head * DHEAD + dr) * SEQ + key0 + kc * 8) * 2);
        async_b128(ldsV + dr * 80 + kc * 16, voff, vt);
      }
    }
    asm volatile("s_wait_asynccnt 0" ::: "memory");
    __syncthreads();

    // scores: two 16x16 tiles, K-dim = 128 in 4 wmma steps
    v8f sacc[2];
#pragma unroll
    for (int n = 0; n < 2; ++n) {
      v8f acc = {};
      const unsigned short* kr = lds_k[n * 16 + row16];   // B col = key
#pragma unroll
      for (int kk = 0; kk < 4; ++kk) {
        BFrag bf;
#pragma unroll
        for (int vg = 0; vg < 8; ++vg) {
          int d = kk * 32 + ((vg >= 4) ? 16 : 0) + hb * 8 + (vg & 3) * 2;  // B layout K pairs (table order)
          bf.u[vg] = *(const unsigned int*)&kr[d];
        }
        acc = wmma_bf16(qf[kk].v, bf.v, acc);
      }
      sacc[n] = acc;
    }

    // causal mask + online softmax
#pragma unroll
    for (int vg = 0; vg < 8; ++vg) {
      int rowg = q0 + hb * 8 + vg;
      float s0 = sacc[0][vg]; if (key0 + row16      > rowg) s0 = -1e30f;
      float s1 = sacc[1][vg]; if (key0 + 16 + row16 > rowg) s1 = -1e30f;
      float m = fmaxf(s0, s1);
#pragma unroll
      for (int off = 1; off < 16; off <<= 1) m = fmaxf(m, __shfl_xor(m, off, 32));
      float m_new = fmaxf(m_prev[vg], m);
      float scale = __expf(m_prev[vg] - m_new);
      float p0 = __expf(s0 - m_new);
      float p1 = __expf(s1 - m_new);
      float rs = p0 + p1;
#pragma unroll
      for (int off = 1; off < 16; off <<= 1) rs += __shfl_xor(rs, off, 32);
      l_run[vg]  = l_run[vg] * scale + rs;
      m_prev[vg] = m_new;
#pragma unroll
      for (int t = 0; t < 8; ++t) ctx_acc[t][vg] *= scale;
      lds_p[hb * 8 + vg][row16]      = f2bf(p0);
      lds_p[hb * 8 + vg][16 + row16] = f2bf(p1);
    }
    __syncthreads();

    // P as one exact 16x32 A fragment
    BFrag pf;
#pragma unroll
    for (int vg = 0; vg < 8; ++vg) {
      int kx = ((vg >= 4) ? 16 : 0) + hb * 8 + (vg & 3) * 2;
      pf.u[vg] = *(const unsigned int*)&lds_p[row16][kx];
    }
    // ctx += P (16x32) x V (32x128)
#pragma unroll
    for (int t = 0; t < 8; ++t) {
      BFrag bf;
      const unsigned short* vr = lds_vt[t * 16 + row16];  // B col = d
#pragma unroll
      for (int vg = 0; vg < 8; ++vg) {
        int ky = ((vg >= 4) ? 16 : 0) + hb * 8 + (vg & 3) * 2;  // key pair (table order)
        bf.u[vg] = *(const unsigned int*)&vr[ky];
      }
      ctx_acc[t] = wmma_bf16(pf.v, bf.v, ctx_acc[t]);
    }
    __syncthreads();
  }

  // normalize and write ctx bf16 [S][MODELD]
#pragma unroll
  for (int vg = 0; vg < 8; ++vg) {
    float inv = 1.0f / l_run[vg];
    unsigned short* orow =
        ctx_h + (size_t)(q0 + hb * 8 + vg) * MODELD + head * DHEAD;
#pragma unroll
    for (int t = 0; t < 8; ++t) orow[t * 16 + row16] = f2bf(ctx_acc[t][vg] * inv);
  }
}

// ---------------- GEMM: epi(A[M,K]bf16 @ W[N,K]bf16^T + bias) ----------------
// Tiles staged by the Tensor Data Mover (wave 0 issues descriptors), double-buffered.
// MODE 1: + coef*res   MODE 2: bias only   MODE 3: exact gelu
template <int MODE, bool WF, bool WH>
__global__ __launch_bounds__(256)
void gemm_kernel(const unsigned short* __restrict__ A,
                 const unsigned short* __restrict__ W,
                 const float* __restrict__ bias, const float* __restrict__ res,
                 float* __restrict__ outf, unsigned short* __restrict__ outh,
                 int M, int N, int K, float coef) {
  __shared__ unsigned short lds_a[2][128][40];   // 128x32 bf16, 80B rows (TDM-padded)
  __shared__ unsigned short lds_b[2][64][40];    //  64x32 bf16, 80B rows

  const int tid   = threadIdx.x;
  const int lane  = tid & 31;
  const int wv    = tid >> 5;
  const int wr    = wv & 3;       // wave row (4)
  const int wc    = wv >> 2;      // wave col (2)
  const int row16 = lane & 15;
  const int hb    = lane >> 4;
  const int m0    = blockIdx.y * 128;
  const int n0    = blockIdx.x * 64;

  const unsigned ldsA[2] = {(unsigned)(size_t)&lds_a[0][0][0],
                            (unsigned)(size_t)&lds_a[1][0][0]};
  const unsigned ldsB[2] = {(unsigned)(size_t)&lds_b[0][0][0],
                            (unsigned)(size_t)&lds_b[1][0][0]};

  v8f acc[2][2];
  acc[0][0] = {}; acc[0][1] = {}; acc[1][0] = {}; acc[1][1] = {};

  const int ntiles = K / 32;
  if (wv == 0) {                                   // prologue: tile 0 -> buf 0
    tdm_tile_2d(A, (long long)m0 * K, K, M, 128, ldsA[0]);
    tdm_tile_2d(W, (long long)n0 * K, K, N, 64,  ldsB[0]);
  }

  int buf = 0;
  for (int t = 0; t < ntiles; ++t) {
    if (wv == 0) {
      if (t + 1 < ntiles) {                        // prefetch next tile -> other buf
        long long kk0 = (long long)(t + 1) * 32;
        tdm_tile_2d(A, (long long)m0 * K + kk0, K, M, 128, ldsA[buf ^ 1]);
        tdm_tile_2d(W, (long long)n0 * K + kk0, K, N, 64,  ldsB[buf ^ 1]);
        __builtin_amdgcn_s_wait_tensorcnt(2);      // tile t landed
      } else {
        __builtin_amdgcn_s_wait_tensorcnt(0);
      }
    }
    __syncthreads();

    BFrag af[2], bf[2];
#pragma unroll
    for (int i = 0; i < 2; ++i) {
      int arow = wr * 32 + i * 16 + row16;
      int brow = wc * 32 + i * 16 + row16;
#pragma unroll
      for (int vg = 0; vg < 8; ++vg) {
        int ka = ((vg >= 4) ? 16 : 0) + hb * 8 + (vg & 3) * 2;  // A layout
        int kb = ((vg >= 4) ? 16 : 0) + hb * 8 + (vg & 3) * 2;  // B layout (table order)
        af[i].u[vg] = *(const unsigned int*)&lds_a[buf][arow][ka];
        bf[i].u[vg] = *(const unsigned int*)&lds_b[buf][brow][kb];
      }
    }
#pragma unroll
    for (int i = 0; i < 2; ++i)
#pragma unroll
      for (int j = 0; j < 2; ++j)
        acc[i][j] = wmma_bf16(af[i].v, bf[j].v, acc[i][j]);
    __syncthreads();
    buf ^= 1;
  }

  // epilogue
#pragma unroll
  for (int i = 0; i < 2; ++i)
#pragma unroll
    for (int j = 0; j < 2; ++j)
#pragma unroll
      for (int vg = 0; vg < 8; ++vg) {
        int r = m0 + wr * 32 + i * 16 + hb * 8 + vg;
        int c = n0 + wc * 32 + j * 16 + row16;
        size_t idx = (size_t)r * N + c;
        float x = acc[i][j][vg] + bias[c];
        if (MODE == 1) x += coef * res[idx];
        if (MODE == 3) x = 0.5f * x * (1.0f + erff(x * 0.70710678118654752f));
        if (WF) outf[idx] = x;
        if (WH) outh[idx] = f2bf(x);
      }
}

// ---------------- LayerNorm: one block per row ----------------
template <bool BF16OUT>
__global__ __launch_bounds__(256)
void ln_kernel(const float* __restrict__ in, float* __restrict__ outf,
               unsigned short* __restrict__ outh, int N) {
  __shared__ float rs[8], rs2[8];
  const int row = blockIdx.x;
  const float* x = in + (size_t)row * N;
  float s = 0.f, s2 = 0.f;
  for (int c = threadIdx.x; c < N; c += 256) { float v = x[c]; s += v; s2 += v * v; }
#pragma unroll
  for (int off = 1; off < 32; off <<= 1) {
    s  += __shfl_xor(s,  off, 32);
    s2 += __shfl_xor(s2, off, 32);
  }
  if ((threadIdx.x & 31) == 0) { rs[threadIdx.x >> 5] = s; rs2[threadIdx.x >> 5] = s2; }
  __syncthreads();
  if (threadIdx.x == 0) {
    float a = 0.f, b = 0.f;
    for (int i = 0; i < 8; ++i) { a += rs[i]; b += rs2[i]; }
    rs[0] = a; rs2[0] = b;
  }
  __syncthreads();
  float mean = rs[0] / N;
  float var  = rs2[0] / N - mean * mean;
  float inv  = rsqrtf(var + 1e-5f);
  for (int c = threadIdx.x; c < N; c += 256) {
    float y = (x[c] - mean) * inv;
    if (BF16OUT) outh[(size_t)row * N + c] = f2bf(y);
    else         outf[(size_t)row * N + c] = y;
  }
}

// ---------------- orchestration ----------------
extern "C" void kernel_launch(void* const* d_in, const int* in_sizes, int n_in,
                              void* d_out, int out_size, void* d_ws, size_t ws_size,
                              hipStream_t stream) {
  (void)in_sizes; (void)n_in; (void)out_size; (void)ws_size;
  const float* q      = (const float*)d_in[0];
  const float* k      = (const float*)d_in[1];
  const float* v      = (const float*)d_in[2];
  // d_in[3] = attention_mask (causal, implemented analytically)
  const float* resid  = (const float*)d_in[4];
  const float* W_attn = (const float*)d_in[5];
  const float* b_attn = (const float*)d_in[6];
  const float* W_post = (const float*)d_in[7];
  const float* b_post = (const float*)d_in[8];
  const float* W_mlp1 = (const float*)d_in[9];
  const float* b_mlp1 = (const float*)d_in[10];
  const float* W_mlp2 = (const float*)d_in[11];
  const float* b_mlp2 = (const float*)d_in[12];
  float* out = (float*)d_out;

  // workspace carve-out
  char* p = (char*)d_ws;
  auto alloc = [&](size_t bytes) { char* r = p; p += (bytes + 255) & ~(size_t)255; return r; };
  const size_t SM = (size_t)SEQ * MODELD;
  const size_t SF = (size_t)SEQ * FFD;
  unsigned short* qh       = (unsigned short*)alloc(SM * 2);
  unsigned short* kh       = (unsigned short*)alloc(SM * 2);
  unsigned short* vtb      = (unsigned short*)alloc(SM * 2);
  unsigned short* ctx_h    = (unsigned short*)alloc(SM * 2);
  unsigned short* wattn_h  = (unsigned short*)alloc((size_t)MODELD * MODELD * 2);
  unsigned short* wpost_h  = (unsigned short*)alloc((size_t)MODELD * MODELD * 2);
  unsigned short* wmlp1_h  = (unsigned short*)alloc((size_t)FFD * MODELD * 2);
  unsigned short* wmlp2_h  = (unsigned short*)alloc((size_t)MODELD * FFD * 2);
  unsigned short* normed_h = (unsigned short*)alloc(SM * 2);
  unsigned short* affine_h = (unsigned short*)alloc(SM * 2);
  unsigned short* h_h      = (unsigned short*)alloc(SF * 2);
  float*          attn_out = (float*)alloc(SM * 4);
  float*          affine   = (float*)alloc(SM * 4);
  float*          mlp_out  = (float*)alloc(SM * 4);

  // one-shot bf16 conversions (weights reused 16x per GEMM; bf16 fits L2)
  auto cvt = [&](const float* src, unsigned short* dst, size_t n, float scale) {
    cvt_bf16_kernel<<<(unsigned)(n / 4 / 256), 256, 0, stream>>>(src, dst, n / 4, scale);
  };
  cvt(q, qh, SM, QSCALE);
  cvt(k, kh, SM, 1.0f);
  cvt_vt_kernel<<<(unsigned)(SM / 256), 256, 0, stream>>>(v, vtb);
  cvt(W_attn, wattn_h, (size_t)MODELD * MODELD, 1.0f);
  cvt(W_post, wpost_h, (size_t)MODELD * MODELD, 1.0f);
  cvt(W_mlp1, wmlp1_h, (size_t)FFD * MODELD, 1.0f);
  cvt(W_mlp2, wmlp2_h, (size_t)MODELD * FFD, 1.0f);

  // attention -> ctx (bf16)
  attn_kernel<<<dim3(SEQ / 16, HEADS), 32, 0, stream>>>(qh, kh, vtb, ctx_h);
  // attn_out = ctx @ W_attn^T + b_attn + RESC*residual   (f32)
  gemm_kernel<1, true, false><<<dim3(MODELD / 64, SEQ / 128), 256, 0, stream>>>(
      ctx_h, wattn_h, b_attn, resid, attn_out, nullptr, SEQ, MODELD, MODELD, RESC);
  // normed = LN(attn_out)  (bf16)
  ln_kernel<true><<<SEQ, 256, 0, stream>>>(attn_out, nullptr, normed_h, MODELD);
  // affine = normed @ W_post^T + b_post  (f32 + bf16)
  gemm_kernel<2, true, true><<<dim3(MODELD / 64, SEQ / 128), 256, 0, stream>>>(
      normed_h, wpost_h, b_post, nullptr, affine, affine_h, SEQ, MODELD, MODELD, 0.f);
  // h = gelu(affine @ W_mlp1^T + b_mlp1)  (bf16)
  gemm_kernel<3, false, true><<<dim3(FFD / 64, SEQ / 128), 256, 0, stream>>>(
      affine_h, wmlp1_h, b_mlp1, nullptr, nullptr, h_h, SEQ, FFD, MODELD, 0.f);
  // mlp_out = h @ W_mlp2^T + b_mlp2 + RESC*affine  (f32)
  gemm_kernel<1, true, false><<<dim3(MODELD / 64, SEQ / 128), 256, 0, stream>>>(
      h_h, wmlp2_h, b_mlp2, affine, mlp_out, nullptr, SEQ, MODELD, FFD, RESC);
  // out = LN(mlp_out)  (f32)
  ln_kernel<false><<<SEQ, 256, 0, stream>>>(mlp_out, out, nullptr, MODELD);
}
